// EncoderLayer_45552423141995
// MI455X (gfx1250) — matrix operand order repaired and checked by reading, not compiled, not run on hardware
//
#include <hip/hip_runtime.h>

typedef _Float16 half_t;
typedef __attribute__((ext_vector_type(8)))  _Float16 v8h;
typedef __attribute__((ext_vector_type(16))) _Float16 v16h;
typedef __attribute__((ext_vector_type(8)))  float    v8f;
typedef __attribute__((ext_vector_type(4)))  unsigned u32x4;
typedef __attribute__((ext_vector_type(8)))  unsigned u32x8;

// Problem constants
static constexpr int Bb  = 2;
static constexpr int Ss  = 2048;
static constexpr int Dd  = 1024;
static constexpr int Hh  = 16;
static constexpr int Ff  = 4096;
static constexpr int DHh = 64;
static constexpr int Mm  = Bb * Ss;   // 4096 rows

static __device__ inline v16h cat8(v8h a, v8h b) {
  return __builtin_shufflevector(a, b, 0,1,2,3,4,5,6,7,8,9,10,11,12,13,14,15);
}

static __device__ inline v8f wmma_f16(v16h a, v16h b, v8f c) {
  // D = A(16x32 f16) * B(32x16 f16) + C(16x16 f32)
  return __builtin_amdgcn_wmma_f32_16x16x32_f16(false, a, false, b, (short)0, c,
                                                false, false);
}

// ---------------------------------------------------------------------------
// TDM: 2D tile copy global -> LDS, element = DWORD, with LDS row padding.
// Descriptor per CDNA5 ISA. Issued by one wave,
// tracked with TENSORcnt.
//   tile: tile_rows rows of row_dwords DWORDs; global row pitch stride_dwords
//   LDS : rows stored contiguously + (pad_amount) after every (pad_interval)
// ---------------------------------------------------------------------------
static __device__ inline void tdm_load_2d(unsigned lds_addr, const void* gptr,
                                          unsigned tile_rows, unsigned row_dwords,
                                          unsigned stride_dwords,
                                          unsigned pad_interval_code,   // 3 => 16 DW
                                          unsigned pad_amount_code) {   // 7 => 8 DW
  unsigned long long ga = (unsigned long long)(uintptr_t)gptr;
  u32x4 g0;
  g0[0] = 1u;                                        // count=1, user descriptor
  g0[1] = lds_addr;                                  // lds_addr [63:32]
  g0[2] = (unsigned)(ga & 0xFFFFFFFFu);              // global_addr[31:0]
  g0[3] = (unsigned)((ga >> 32) & 0x01FFFFFFu)       // global_addr[56:32]
        | (2u << 30);                                // type = 2 ("image")
  u32x8 g1;
  g1[0] = (2u << 16)                                 // data_size = 4B
        | (1u << 20)                                 // pad_enable
        | (pad_interval_code << 22)
        | (pad_amount_code << 25);
  g1[1] = (row_dwords & 0xFFFFu) << 16;                               // tensor_dim0 lo
  g1[2] = ((row_dwords >> 16) & 0xFFFFu) | ((tile_rows & 0xFFFFu) << 16); // dim0 hi | tensor_dim1 lo
  g1[3] = ((tile_rows >> 16) & 0xFFFFu) | ((row_dwords & 0xFFFFu) << 16); // dim1 hi | tile_dim0
  g1[4] = (tile_rows & 0xFFFFu);                                      // tile_dim1 (tile_dim2=0)
  g1[5] = stride_dwords;                                              // tensor_dim0_stride lo32
  g1[6] = 0u;                                                         // stride hi | dim1_stride lo
  g1[7] = 0u;
  asm volatile("tensor_load_to_lds %0, %1" :: "s"(g0), "s"(g1) : "memory");
}

// Per-lane async global->LDS 16B copy (ASYNCcnt).
static __device__ inline void async_copy16(void* lds_dst, const void* gsrc) {
  unsigned lo = (unsigned)(uintptr_t)lds_dst;
  unsigned long long ga = (unsigned long long)(uintptr_t)gsrc;
  asm volatile("global_load_async_to_lds_b128 %0, %1, off"
               :: "v"(lo), "v"(ga) : "memory");
}

// ============================================================================
// 1) x = x_enc + x_pos   (f32 for residual + f16 for WMMA GEMMs)
// ============================================================================
__global__ __launch_bounds__(256) void add_kernel(const float* __restrict__ a,
                                                  const float* __restrict__ b,
                                                  float* __restrict__ o32,
                                                  half_t* __restrict__ o16) {
  int i = blockIdx.x * 256 + threadIdx.x;
  float v = a[i] + b[i];
  o32[i] = v;
  o16[i] = (half_t)v;
}

// ============================================================================
// 2) Transpose + cast: W[K,N] f32  ->  WT[N,K] f16
// ============================================================================
__global__ __launch_bounds__(256) void trcast_kernel(const float* __restrict__ W,
                                                     half_t* __restrict__ WT,
                                                     int K, int N) {
  __shared__ float tile[32][33];
  int n0 = blockIdx.x * 32;
  int k0 = blockIdx.y * 32;
#pragma unroll
  for (int j = 0; j < 4; ++j) {
    int kk = threadIdx.y + j * 8;
    tile[kk][threadIdx.x] = W[(size_t)(k0 + kk) * N + n0 + threadIdx.x];
  }
  __syncthreads();
#pragma unroll
  for (int j = 0; j < 4; ++j) {
    int nn = threadIdx.y + j * 8;
    WT[(size_t)(n0 + nn) * K + k0 + threadIdx.x] = (half_t)tile[threadIdx.x][nn];
  }
}

// ============================================================================
// 3) WMMA GEMM: C[M,N] = A[M,K](f16) * WT[N,K]^T + bias, block tile 64x256,
//    8 waves (2Mx4N), wave tile 32x64, K-step 32. A and WT tiles are staged
//    into padded LDS (96B row pitch) by the Tensor Data Mover: wave0 moves
//    the 256x32 WT tile, wave1 the 64x32 A tile, one TDM instruction each.
// ============================================================================
template <bool RELU, bool OF32, bool OF16>
__global__ __launch_bounds__(256) void gemm_kernel(const half_t* __restrict__ A,
                                                   const half_t* __restrict__ WT,
                                                   const float* __restrict__ bias,
                                                   float* __restrict__ outF,
                                                   half_t* __restrict__ outH,
                                                   int Mdim, int Ndim, int Kdim) {
  __shared__ half_t As[64][48];    // 96B row pitch (64B data + 32B pad)
  __shared__ half_t Ws[256][48];

  const int t = threadIdx.x;
  const int lane = t & 31, wid = t >> 5;
  const int wm = wid & 1, wn = wid >> 1;
  const int m0 = blockIdx.x * 64;
  const int n0 = blockIdx.y * 256;

  v8f zero = {0.f,0.f,0.f,0.f,0.f,0.f,0.f,0.f};
  v8f acc[2][4];
#pragma unroll
  for (int mt = 0; mt < 2; ++mt)
#pragma unroll
    for (int nt = 0; nt < 4; ++nt) acc[mt][nt] = zero;

  const unsigned strideDW = (unsigned)(Kdim / 2);
  const int nk = Kdim / 32;
  for (int kt = 0; kt < nk; ++kt) {
    const int k0 = kt * 32;
    __syncthreads();                       // previous tile fully consumed
    if (wid == 0) {
      tdm_load_2d((unsigned)(uintptr_t)&Ws[0][0],
                  WT + (size_t)n0 * Kdim + k0, 256u, 16u, strideDW, 3u, 7u);
      __builtin_amdgcn_s_wait_tensorcnt((short)0);
    } else if (wid == 1) {
      tdm_load_2d((unsigned)(uintptr_t)&As[0][0],
                  A + (size_t)m0 * Kdim + k0, 64u, 16u, strideDW, 3u, 7u);
      __builtin_amdgcn_s_wait_tensorcnt((short)0);
    }
    __syncthreads();                       // tiles visible to all waves

    v16h afrag[2];
#pragma unroll
    for (int mt = 0; mt < 2; ++mt) {
      int m = wm * 32 + mt * 16 + (lane & 15);
      int kb = (lane >> 4) * 8;
      afrag[mt] = cat8(*(const v8h*)&As[m][kb], *(const v8h*)&As[m][kb + 16]);
    }
#pragma unroll
    for (int nt = 0; nt < 4; ++nt) {
      int n = wn * 64 + nt * 16 + (lane & 15);
      int kb = (lane >> 4) * 8;
      v16h bfrag = cat8(*(const v8h*)&Ws[n][kb], *(const v8h*)&Ws[n][kb + 16]);
#pragma unroll
      for (int mt = 0; mt < 2; ++mt)
        acc[mt][nt] = wmma_f16(afrag[mt], bfrag, acc[mt][nt]);
    }
  }

  // epilogue: C layout -> VGPR r holds row (r + 8*(lane>=16)), col = lane&15
  const int chi = lane >> 4, nc = lane & 15;
#pragma unroll
  for (int mt = 0; mt < 2; ++mt) {
#pragma unroll
    for (int nt = 0; nt < 4; ++nt) {
      int col = n0 + wn * 64 + nt * 16 + nc;
      float bv = bias[col];
#pragma unroll
      for (int r = 0; r < 8; ++r) {
        int row = m0 + wm * 32 + mt * 16 + r + 8 * chi;
        float c = acc[mt][nt][r] + bv;
        if (RELU) c = fmaxf(c, 0.0f);
        size_t idx = (size_t)row * Ndim + col;
        if (OF32) outF[idx] = c;
        if (OF16) outH[idx] = (half_t)c;
      }
    }
  }
}

// ============================================================================
// 4) Flash attention: one block per (b, h, 128-query tile); 8 waves, each
//    wave owns 16 queries. Key blocks of 32: K staged by async global->LDS
//    (ASYNCcnt), V staged manually transposed; online softmax in registers.
//    Output written directly in f16 for the projection GEMM.
// ============================================================================
__global__ __launch_bounds__(256) void attn_kernel(const half_t* __restrict__ q,
                                                   const half_t* __restrict__ k,
                                                   const half_t* __restrict__ v,
                                                   half_t* __restrict__ o) {
  __shared__ half_t Ks[32][72];        // [key][d]   (144B row pitch)
  __shared__ half_t Vt[64][48];        // [d][key]   (96B row pitch)
  __shared__ half_t Ps[8][16][48];     // per-wave P tile [q][key]

  const int t = threadIdx.x, lane = t & 31, w = t >> 5;
  const size_t base = ((size_t)blockIdx.z * Ss) * Dd + (size_t)blockIdx.y * DHh;
  const int q0 = blockIdx.x * 128 + w * 16;

  // Q fragments (16 x 64 -> two 16x32 A-fragments), loaded once from global
  v16h qf[2];
  {
    int s  = q0 + (lane & 15);
    int kb = (lane >> 4) * 8;
    const half_t* qp = q + base + (size_t)s * Dd;
#pragma unroll
    for (int c = 0; c < 2; ++c)
      qf[c] = cat8(*(const v8h*)(qp + c * 32 + kb),
                   *(const v8h*)(qp + c * 32 + kb + 16));
  }

  v8f zero = {0.f,0.f,0.f,0.f,0.f,0.f,0.f,0.f};
  float mrow[8], lrow[8];
  v8f oacc[4];
#pragma unroll
  for (int r = 0; r < 8; ++r) { mrow[r] = -1e30f; lrow[r] = 0.0f; }
#pragma unroll
  for (int nt = 0; nt < 4; ++nt) oacc[nt] = zero;

  for (int kb0 = 0; kb0 < Ss / 32; ++kb0) {
    const int key0 = kb0 * 32;
    __syncthreads();
    {   // stage: K block via async copy, V block transposed manually
      int krow = t >> 3;
      int cb   = (t & 7) * 8;
      async_copy16(&Ks[krow][cb], k + base + (size_t)(key0 + krow) * Dd + cb);
      const half_t* vp = v + base + (size_t)(key0 + krow) * Dd + cb;
      v8h vv = *(const v8h*)vp;
#pragma unroll
      for (int j = 0; j < 8; ++j) Vt[cb + j][krow] = vv[j];
    }
    asm volatile("s_wait_asynccnt 0" ::: "memory");
    __syncthreads();

    // logits: two 16(q) x 16(key) tiles, K-dim = DH = 64 (2 WMMA steps each)
    v8f lg0 = zero, lg1 = zero;
#pragma unroll
    for (int c = 0; c < 2; ++c) {
      int kb2 = c * 32 + (lane >> 4) * 8;
      int kl  = lane & 15;
      v16h b0 = cat8(*(const v8h*)&Ks[kl][kb2],      *(const v8h*)&Ks[kl][kb2 + 16]);
      v16h b1 = cat8(*(const v8h*)&Ks[16 + kl][kb2], *(const v8h*)&Ks[16 + kl][kb2 + 16]);
      lg0 = wmma_f16(qf[c], b0, lg0);
      lg1 = wmma_f16(qf[c], b1, lg1);
    }

    // online softmax update (row r lives in VGPR r, cols across 16-lane half)
#pragma unroll
    for (int r = 0; r < 8; ++r) {
      float l0 = lg0[r] * 0.125f, l1 = lg1[r] * 0.125f;
      float mx = fmaxf(l0, l1);
#pragma unroll
      for (int off = 1; off < 16; off <<= 1)
        mx = fmaxf(mx, __shfl_xor(mx, off, 32));
      float mnew = fmaxf(mrow[r], mx);
      float p0 = __expf(l0 - mnew), p1 = __expf(l1 - mnew);
      float ps = p0 + p1;
#pragma unroll
      for (int off = 1; off < 16; off <<= 1) ps += __shfl_xor(ps, off, 32);
      float alpha = __expf(mrow[r] - mnew);
      lrow[r] = lrow[r] * alpha + ps;
      mrow[r] = mnew;
#pragma unroll
      for (int nt = 0; nt < 4; ++nt) oacc[nt][r] *= alpha;
      int prow = r + 8 * (lane >> 4);
      Ps[w][prow][lane & 15]        = (half_t)p0;
      Ps[w][prow][16 + (lane & 15)] = (half_t)p1;
    }
    asm volatile("s_wait_dscnt 0" ::: "memory");  // wave-private LDS RAW

    // P (16x32) A-fragment, then P*V onto 4 accumulators (d chunks of 16)
    v16h pf;
    {
      int prow = lane & 15;
      int kb = (lane >> 4) * 8;
      pf = cat8(*(const v8h*)&Ps[w][prow][kb], *(const v8h*)&Ps[w][prow][kb + 16]);
    }
#pragma unroll
    for (int nt = 0; nt < 4; ++nt) {
      int dl  = nt * 16 + (lane & 15);
      int kb2 = (lane >> 4) * 8;
      v16h bv = cat8(*(const v8h*)&Vt[dl][kb2], *(const v8h*)&Vt[dl][kb2 + 16]);
      oacc[nt] = wmma_f16(pf, bv, oacc[nt]);
    }
  }

  {   // normalize by l and store f16 (input to projection GEMM)
    int chi = lane >> 4, nc = lane & 15;
#pragma unroll
    for (int r = 0; r < 8; ++r) {
      float inv = 1.0f / lrow[r];
      int s = q0 + r + 8 * chi;
#pragma unroll
      for (int nt = 0; nt < 4; ++nt)
        o[base + (size_t)s * Dd + nt * 16 + nc] = (half_t)(oacc[nt][r] * inv);
    }
  }
}

// ============================================================================
// 5) out = res + LayerNorm(y) * g + be   (optional f16 copy of the output)
// ============================================================================
template <bool OUTH>
__global__ __launch_bounds__(256) void ln_kernel(const float* __restrict__ y,
                                                 const float* __restrict__ res,
                                                 const float* __restrict__ g,
                                                 const float* __restrict__ be,
                                                 float* __restrict__ outF,
                                                 half_t* __restrict__ outH) {
  __shared__ float sred[8], s2red[8];
  const size_t rb = (size_t)blockIdx.x * Dd;
  float s = 0.f, s2 = 0.f;
#pragma unroll
  for (int j = 0; j < 4; ++j) {
    float tv = y[rb + threadIdx.x + j * 256];
    s += tv; s2 += tv * tv;
  }
#pragma unroll
  for (int off = 16; off > 0; off >>= 1) {
    s  += __shfl_xor(s,  off, 32);
    s2 += __shfl_xor(s2, off, 32);
  }
  int lane = threadIdx.x & 31, w = threadIdx.x >> 5;
  if (lane == 0) { sred[w] = s; s2red[w] = s2; }
  __syncthreads();
  float tot = 0.f, tot2 = 0.f;
#pragma unroll
  for (int j = 0; j < 8; ++j) { tot += sred[j]; tot2 += s2red[j]; }
  float mu  = tot * (1.0f / Dd);
  float var = tot2 * (1.0f / Dd) - mu * mu;
  float rs  = rsqrtf(var + 1e-6f);
#pragma unroll
  for (int j = 0; j < 4; ++j) {
    int i = threadIdx.x + j * 256;
    float tv = y[rb + i];
    float o = res[rb + i] + (tv - mu) * rs * g[i] + be[i];
    outF[rb + i] = o;
    if (OUTH) outH[rb + i] = (half_t)o;
  }
}

// ============================================================================
extern "C" void kernel_launch(void* const* d_in, const int* in_sizes, int n_in,
                              void* d_out, int out_size, void* d_ws, size_t ws_size,
                              hipStream_t stream) {
  const float* x_enc = (const float*)d_in[0];
  const float* x_pos = (const float*)d_in[1];
  const float* wq = (const float*)d_in[2];
  const float* bq = (const float*)d_in[3];
  const float* wk = (const float*)d_in[4];
  const float* bk = (const float*)d_in[5];
  const float* wv = (const float*)d_in[6];
  const float* bv = (const float*)d_in[7];
  const float* wc = (const float*)d_in[8];
  const float* bc = (const float*)d_in[9];
  const float* w1 = (const float*)d_in[10];
  const float* b1 = (const float*)d_in[11];
  const float* w2 = (const float*)d_in[12];
  const float* b2 = (const float*)d_in[13];
  const float* g1  = (const float*)d_in[14];
  const float* be1 = (const float*)d_in[15];
  const float* g2  = (const float*)d_in[16];
  const float* be2 = (const float*)d_in[17];
  float* out = (float*)d_out;

  char* ws = (char*)d_ws;
  const size_t SZ_F32 = (size_t)Mm * Dd * 4;   // 16 MB
  const size_t SZ_F16 = (size_t)Mm * Dd * 2;   //  8 MB
  size_t off = 0;
  float*  x_f32  = (float*)(ws + off);  off += SZ_F32;
  half_t* x_f16  = (half_t*)(ws + off); off += SZ_F16;
  half_t* qh     = (half_t*)(ws + off); off += SZ_F16;
  half_t* kh     = (half_t*)(ws + off); off += SZ_F16;
  half_t* vh     = (half_t*)(ws + off); off += SZ_F16;
  half_t* attn_h = (half_t*)(ws + off); off += SZ_F16;
  float*  proj_f = (float*)(ws + off);  off += SZ_F32;   // reused as ffn
  float*  a_f32  = (float*)(ws + off);  off += SZ_F32;
  half_t* a_f16  = (half_t*)(ws + off); off += SZ_F16;
  half_t* h_h    = (half_t*)(ws + off); off += (size_t)Mm * Ff * 2;   // 32 MB
  half_t* wqT    = (half_t*)(ws + off); off += (size_t)Dd * Dd * 2;
  half_t* wkT    = (half_t*)(ws + off); off += (size_t)Dd * Dd * 2;
  half_t* wvT    = (half_t*)(ws + off); off += (size_t)Dd * Dd * 2;
  half_t* wcT    = (half_t*)(ws + off); off += (size_t)Dd * Dd * 2;
  half_t* w1T    = (half_t*)(ws + off); off += (size_t)Dd * Ff * 2;
  half_t* w2T    = (half_t*)(ws + off); off += (size_t)Ff * Dd * 2;
  float*  ffn_f  = proj_f;  // proj dead after ln1

  // 1) x = x_enc + x_pos
  add_kernel<<<(Mm * Dd) / 256, 256, 0, stream>>>(x_enc, x_pos, x_f32, x_f16);

  // 2) weight transpose + cast
  dim3 tb(32, 8);
  trcast_kernel<<<dim3(Dd / 32, Dd / 32), tb, 0, stream>>>(wq, wqT, Dd, Dd);
  trcast_kernel<<<dim3(Dd / 32, Dd / 32), tb, 0, stream>>>(wk, wkT, Dd, Dd);
  trcast_kernel<<<dim3(Dd / 32, Dd / 32), tb, 0, stream>>>(wv, wvT, Dd, Dd);
  trcast_kernel<<<dim3(Dd / 32, Dd / 32), tb, 0, stream>>>(wc, wcT, Dd, Dd);
  trcast_kernel<<<dim3(Ff / 32, Dd / 32), tb, 0, stream>>>(w1, w1T, Dd, Ff);
  trcast_kernel<<<dim3(Dd / 32, Ff / 32), tb, 0, stream>>>(w2, w2T, Ff, Dd);

  // 3) Q, K, V projections (f16 outputs)
  dim3 gqkv(Mm / 64, Dd / 256);
  gemm_kernel<false, false, true><<<gqkv, 256, 0, stream>>>(
      x_f16, wqT, bq, nullptr, qh, Mm, Dd, Dd);
  gemm_kernel<false, false, true><<<gqkv, 256, 0, stream>>>(
      x_f16, wkT, bk, nullptr, kh, Mm, Dd, Dd);
  gemm_kernel<false, false, true><<<gqkv, 256, 0, stream>>>(
      x_f16, wvT, bv, nullptr, vh, Mm, Dd, Dd);

  // 4) flash attention -> attn_h (f16, [B,S,D])
  attn_kernel<<<dim3(Ss / 128, Hh, Bb), 256, 0, stream>>>(qh, kh, vh, attn_h);

  // 5) output projection (f32 out)
  gemm_kernel<false, true, false><<<gqkv, 256, 0, stream>>>(
      attn_h, wcT, bc, proj_f, nullptr, Mm, Dd, Dd);

  // 6) a = x + LN(proj)  (f32 + f16)
  ln_kernel<true><<<Mm, 256, 0, stream>>>(proj_f, x_f32, g1, be1, a_f32, a_f16);

  // 7) FFN1: h = relu(a @ w1 + b1)   (f16 out)
  gemm_kernel<true, false, true><<<dim3(Mm / 64, Ff / 256), 256, 0, stream>>>(
      a_f16, w1T, b1, nullptr, h_h, Mm, Ff, Dd);

  // 8) FFN2: ffn = h @ w2 + b2       (f32 out)
  gemm_kernel<false, true, false><<<dim3(Mm / 64, Dd / 256), 256, 0, stream>>>(
      h_h, w2T, b2, ffn_f, nullptr, Mm, Dd, Ff);

  // 9) out = a + LN(ffn)
  ln_kernel<false><<<Mm, 256, 0, stream>>>(ffn_f, a_f32, g2, be2, out, nullptr);

  (void)in_sizes; (void)n_in; (void)out_size; (void)ws_size;
}
